// CausalSelfAttention_87127706566683
// MI455X (gfx1250) — matrix operand order repaired and checked by reading, not compiled, not run on hardware
//
#include <hip/hip_runtime.h>

// ---------------- problem constants ----------------
#define BB 2
#define TT 2048
#define CC 1024
#define HH 16
#define DD 64
#define MM (BB * TT)          // 4096 rows
#define NEG_BIG -1.0e10f

typedef __attribute__((ext_vector_type(16))) _Float16 v16h;
typedef __attribute__((ext_vector_type(8)))  float    v8f;
typedef __attribute__((ext_vector_type(4)))  _Float16 v4h;
typedef int v4i __attribute__((__vector_size__(4 * sizeof(int))));

#define WMMA_F16(a, b, c) \
    __builtin_amdgcn_wmma_f32_16x16x32_f16(false, (a), false, (b), (short)0, (c), false, false)

// ---------------- async global->LDS copy (CDNA5 ASYNCcnt path) ----------------
#define AS1 __attribute__((address_space(1)))
#define AS3 __attribute__((address_space(3)))

#if defined(__has_builtin)
#if __has_builtin(__builtin_amdgcn_global_load_async_to_lds_b128)
#define HAVE_ASYNC_LDS 1
#endif
#endif
#ifndef HAVE_ASYNC_LDS
#define HAVE_ASYNC_LDS 0
#endif

// copy 16 bytes (8 halves) global -> LDS
static __device__ __forceinline__ void cp16(const _Float16* g, _Float16* l) {
#if HAVE_ASYNC_LDS
    __builtin_amdgcn_global_load_async_to_lds_b128(
        (AS1 v4i*)(const_cast<_Float16*>(g)), (AS3 v4i*)(l), 0, 0);
#else
    *(uint4*)l = *(const uint4*)g;
#endif
}

static __device__ __forceinline__ void async_wait() {
#if HAVE_ASYNC_LDS
    asm volatile("s_wait_asynccnt 0" ::: "memory");
#endif
}

// ---------------- fragment loaders (ISA 7.12.2 layouts) ----------------
// A 16x32 f16: lane&15 = row M; lane>>4 selects K interleave.
static __device__ __forceinline__ v16h load_a_frag(const _Float16* p, int ld) {
    int lane = threadIdx.x & 31;
    const _Float16* q = p + (lane & 15) * ld + ((lane >> 4) << 3);
    v16h a;
    ((uint4*)&a)[0] = *(const uint4*)(q);
    ((uint4*)&a)[1] = *(const uint4*)(q + 16);
    return a;
}

// B 32x16 f16: lane&15 = column N; lane>>4 selects K interleave (same as A).
static __device__ __forceinline__ v16h load_b_frag(const _Float16* p, int ld) {
    int lane = threadIdx.x & 31;
    const _Float16* q = p + (lane & 15) * ld + ((lane >> 4) << 3);
    v16h b;
    ((uint4*)&b)[0] = *(const uint4*)(q);
    ((uint4*)&b)[1] = *(const uint4*)(q + 16);
    return b;
}

static __device__ __forceinline__ v8f vzero8() {
    v8f z = {0.f, 0.f, 0.f, 0.f, 0.f, 0.f, 0.f, 0.f};
    return z;
}

// ---------------- f32 -> f16 convert ----------------
__global__ __launch_bounds__(256) void cvt_f32_f16(const float* __restrict__ in,
                                                   _Float16* __restrict__ out, int n4) {
    int i = blockIdx.x * blockDim.x + threadIdx.x;
    if (i < n4) {
        float4 v = ((const float4*)in)[i];
        v4h h;
        h[0] = (_Float16)v.x; h[1] = (_Float16)v.y;
        h[2] = (_Float16)v.z; h[3] = (_Float16)v.w;
        ((v4h*)out)[i] = h;
    }
}

// ---------------- WMMA GEMM:  Out[M,N] = A[M,K] * Bw[N,K]^T + bias ----------------
// MODE 0: f32 row-major (final projection)
// MODE 1: fused RoPE, f16 store to (B,H,T,D)   (Q, K)
// MODE 2: f16 store to (B,H,T,D)               (V)
// 256 thr (8 waves), tile 128x128, BK=32, double-buffered async LDS pipeline.
template <int MODE>
__global__ __launch_bounds__(256) void gemm_wmma(const _Float16* __restrict__ A,
                                                 const _Float16* __restrict__ Bw,
                                                 const float* __restrict__ bias,
                                                 float* __restrict__ outF,
                                                 _Float16* __restrict__ outH) {
    constexpr int LDT = 40;  // padded LDS row stride (halves), 16B aligned
    __shared__ _Float16 As[2][128 * LDT];
    __shared__ _Float16 Bs[2][128 * LDT];

    const int tid  = threadIdx.x;
    const int lane = tid & 31;
    const int w    = tid >> 5;
    const int wm   = w >> 2;   // 0..1
    const int wn   = w & 3;    // 0..3
    const int m0   = blockIdx.y * 128;
    const int n0   = blockIdx.x * 128;

    auto stage = [&](int kofs, int bufi) {
#pragma unroll
        for (int it = 0; it < 2; ++it) {
            int u   = tid + it * 256;     // 0..511 uint4
            int row = u >> 2;             // 4 uint4 per 32-half row
            int c8  = (u & 3) << 3;
            cp16(A + (size_t)(m0 + row) * CC + kofs + c8, &As[bufi][row * LDT + c8]);
            cp16(Bw + (size_t)(n0 + row) * CC + kofs + c8, &Bs[bufi][row * LDT + c8]);
        }
    };

    v8f acc[4][2];
#pragma unroll
    for (int mi = 0; mi < 4; ++mi)
#pragma unroll
        for (int ni = 0; ni < 2; ++ni) acc[mi][ni] = vzero8();

    stage(0, 0);
    async_wait();
    __syncthreads();

    constexpr int NKT = CC / 32;   // 32 K-tiles
    for (int kt = 0; kt < NKT; ++kt) {
        const int cur = kt & 1;
        if (kt + 1 < NKT) stage((kt + 1) * 32, cur ^ 1);   // async stage of next tile

        v16h af[4];
#pragma unroll
        for (int mi = 0; mi < 4; ++mi)
            af[mi] = load_a_frag(&As[cur][(wm * 64 + mi * 16) * LDT], LDT);
        v16h bf[2];
#pragma unroll
        for (int ni = 0; ni < 2; ++ni)
            bf[ni] = load_b_frag(&Bs[cur][(wn * 32 + ni * 16) * LDT], LDT);
#pragma unroll
        for (int mi = 0; mi < 4; ++mi)
#pragma unroll
            for (int ni = 0; ni < 2; ++ni)
                acc[mi][ni] = WMMA_F16(af[mi], bf[ni], acc[mi][ni]);

        async_wait();
        __syncthreads();
    }

    // epilogue: C/D layout -> lane row = (lane>>4)*8 + r, col = lane&15
#pragma unroll
    for (int mi = 0; mi < 4; ++mi) {
#pragma unroll
        for (int ni = 0; ni < 2; ++ni) {
            int gcol = n0 + wn * 32 + ni * 16 + (lane & 15);
            float bc = bias[gcol];
#pragma unroll
            for (int r = 0; r < 8; ++r) {
                int grow  = m0 + wm * 64 + mi * 16 + ((lane >> 4) << 3) + r;
                float val = acc[mi][ni][r] + bc;
                if (MODE == 0) {
                    outF[(size_t)grow * CC + gcol] = val;
                } else {
                    float res = val;
                    if (MODE == 1) {
                        // RoPE: pair partner sits in the adjacent lane
                        float partner = __shfl_xor(val, 1, 32);
                        int d = gcol & (DD - 1);
                        float ang = (float)(grow & (TT - 1)) *
                                    __powf(10000.0f, (float)(d >> 1) * (1.0f / 32.0f));
                        float sn, cs;
                        __sincosf(ang, &sn, &cs);
                        res = ((d & 1) == 0) ? (val * cs - partner * sn)
                                             : (partner * sn + val * cs);
                    }
                    int bi = grow >> 11;           // / TT
                    int t  = grow & (TT - 1);
                    int hh = gcol >> 6;            // / DD
                    int dd = gcol & (DD - 1);
                    outH[(((size_t)bi * HH + hh) * TT + t) * DD + dd] = (_Float16)res;
                }
            }
        }
    }
}

// ---------------- Flash attention ----------------
// grid (T/64, B*H), 128 thr = 4 waves; wave w owns 16 query rows.
__global__ __launch_bounds__(128) void attn_flash(const _Float16* __restrict__ Q,
                                                  const _Float16* __restrict__ K,
                                                  const _Float16* __restrict__ V,
                                                  _Float16* __restrict__ Y) {
    constexpr int LDK = 72;  // K tile stride (halves)
    constexpr int LDV = 40;  // V^T tile stride
    __shared__ _Float16 Ks[32 * LDK];     // [key][d]
    __shared__ _Float16 Vt[64 * LDV];     // [d][key]  (transposed)
    __shared__ _Float16 Pl[4 * 16 * 32];  // per-wave P tile

    const int tid   = threadIdx.x;
    const int lane  = tid & 31;
    const int w     = tid >> 5;
    const int qtile = blockIdx.x;
    const int bh    = blockIdx.y;
    const int q0    = qtile * 64;
    const int qbase = q0 + w * 16;

    const size_t base = (size_t)bh * TT * DD;
    const _Float16* Qp = Q + base + (size_t)qbase * DD;
    const _Float16* Kp = K + base;
    const _Float16* Vp = V + base;

    // resident Q fragments (d = 0..31, 32..63)
    v16h qa0 = load_a_frag(Qp + 0, DD);
    v16h qa1 = load_a_frag(Qp + 32, DD);

    float m_[8], l_[8];
    v8f o[4];
#pragma unroll
    for (int r = 0; r < 8; ++r) { m_[r] = -3.0e38f; l_[r] = 0.f; }
#pragma unroll
    for (int jd = 0; jd < 4; ++jd) o[jd] = vzero8();

    const int nkt = qtile * 2 + 2;                 // tiles covering keys <= q0+63
    const int ltw = (qbase + 15) >> 5;             // last tile this wave needs

    for (int kt = 0; kt < nkt; ++kt) {
        const int kbase = kt * 32;
        // ---- stage K (async to LDS) and V (register transpose) tiles ----
#pragma unroll
        for (int it = 0; it < 2; ++it) {
            int u   = tid + it * 128;   // 0..255 uint4
            int row = u >> 3;           // 8 uint4 per 64-half row
            int c8  = (u & 7) << 3;
            cp16(Kp + (size_t)(kbase + row) * DD + c8, Ks + row * LDK + c8);
            uint4 vv = *(const uint4*)(Vp + (size_t)(kbase + row) * DD + c8);
            const _Float16* ve = (const _Float16*)&vv;
#pragma unroll
            for (int e = 0; e < 8; ++e) Vt[(c8 + e) * LDV + row] = ve[e];
        }
        if (kt + 1 < nkt) {   // prefetch next K/V tile into L2 (global_prefetch_b8)
            __builtin_prefetch(Kp + (size_t)(kbase + 32 + (tid >> 2)) * DD, 0, 1);
            __builtin_prefetch(Vp + (size_t)(kbase + 32 + (tid >> 2)) * DD, 0, 1);
        }
        async_wait();
        __syncthreads();

        if (kt <= ltw) {
            // ---- S = Q K^T * scale, causal mask ----
            v8f s[2];
#pragma unroll
            for (int j = 0; j < 2; ++j) {
                s[j] = vzero8();
                s[j] = WMMA_F16(qa0, load_b_frag(Ks + (j * 16) * LDK + 0, LDK), s[j]);
                s[j] = WMMA_F16(qa1, load_b_frag(Ks + (j * 16) * LDK + 32, LDK), s[j]);
            }
            float tmax[8];
#pragma unroll
            for (int r = 0; r < 8; ++r) {
                int qrow = qbase + ((lane >> 4) << 3) + r;
#pragma unroll
                for (int j = 0; j < 2; ++j) {
                    int kcol = kbase + j * 16 + (lane & 15);
                    float sv = s[j][r] * 0.125f;           // 1/sqrt(64)
                    s[j][r] = (kcol > qrow) ? NEG_BIG : sv;
                }
                tmax[r] = fmaxf(s[0][r], s[1][r]);
#pragma unroll
                for (int off = 1; off < 16; off <<= 1)
                    tmax[r] = fmaxf(tmax[r], __shfl_xor(tmax[r], off, 32));
            }
            // ---- online softmax ----
            float corr[8], rsum[8];
#pragma unroll
            for (int r = 0; r < 8; ++r) {
                float nm = fmaxf(m_[r], tmax[r]);
                corr[r] = __expf(m_[r] - nm);
                m_[r] = nm;
                float p0 = __expf(s[0][r] - nm);
                float p1 = __expf(s[1][r] - nm);
                s[0][r] = p0; s[1][r] = p1;
                rsum[r] = p0 + p1;
#pragma unroll
                for (int off = 1; off < 16; off <<= 1)
                    rsum[r] += __shfl_xor(rsum[r], off, 32);
                l_[r] = l_[r] * corr[r] + rsum[r];
            }
#pragma unroll
            for (int jd = 0; jd < 4; ++jd)
#pragma unroll
                for (int r = 0; r < 8; ++r) o[jd][r] *= corr[r];

            // ---- re-layout P (C-layout -> A-layout) through LDS ----
            _Float16* Pw = Pl + w * (16 * 32);
#pragma unroll
            for (int j = 0; j < 2; ++j)
#pragma unroll
                for (int r = 0; r < 8; ++r)
                    Pw[(((lane >> 4) << 3) + r) * 32 + j * 16 + (lane & 15)] =
                        (_Float16)s[j][r];
            asm volatile("s_wait_dscnt 0" ::: "memory");
            v16h pa = load_a_frag(Pw, 32);

            // ---- O += P V ----
#pragma unroll
            for (int jd = 0; jd < 4; ++jd)
                o[jd] = WMMA_F16(pa, load_b_frag(Vt + (jd * 16) * LDV, LDV), o[jd]);
        }
        __syncthreads();
    }

    // ---- normalize and write Y in (B,T,C) ----
    const int bi = bh >> 4;
    const int hh = bh & 15;
    float inv[8];
#pragma unroll
    for (int r = 0; r < 8; ++r) inv[r] = 1.0f / l_[r];
#pragma unroll
    for (int jd = 0; jd < 4; ++jd) {
        int dcol = jd * 16 + (lane & 15);
#pragma unroll
        for (int r = 0; r < 8; ++r) {
            int t = qbase + ((lane >> 4) << 3) + r;
            Y[((size_t)bi * TT + t) * CC + hh * DD + dcol] = (_Float16)(o[jd][r] * inv[r]);
        }
    }
}

// ---------------- host launch ----------------
extern "C" void kernel_launch(void* const* d_in, const int* in_sizes, int n_in,
                              void* d_out, int out_size, void* d_ws, size_t ws_size,
                              hipStream_t stream) {
    const float* x  = (const float*)d_in[0];
    const float* Wq = (const float*)d_in[1];
    const float* bq = (const float*)d_in[2];
    const float* Wk = (const float*)d_in[3];
    const float* bk = (const float*)d_in[4];
    const float* Wv = (const float*)d_in[5];
    const float* bv = (const float*)d_in[6];
    const float* Wp = (const float*)d_in[7];
    const float* bp = (const float*)d_in[8];
    float* out = (float*)d_out;

    _Float16* ws = (_Float16*)d_ws;
    const size_t SZ_X = (size_t)MM * CC;      // 4,194,304
    const size_t SZ_W = (size_t)CC * CC;      // 1,048,576
    _Float16* xb = ws;
    _Float16* wq = xb + SZ_X;
    _Float16* wk = wq + SZ_W;
    _Float16* wv = wk + SZ_W;
    _Float16* wp = wv + SZ_W;
    _Float16* qr = wp + SZ_W;                 // (B,H,T,D)
    _Float16* kr = qr + SZ_X;
    _Float16* vr = kr + SZ_X;
    _Float16* yb = vr + SZ_X;                 // (B,T,C)

    // convert inputs to f16
    {
        int n4 = (int)(SZ_X / 4);
        cvt_f32_f16<<<(n4 + 255) / 256, 256, 0, stream>>>(x, xb, n4);
        int w4 = (int)(SZ_W / 4);
        cvt_f32_f16<<<(w4 + 255) / 256, 256, 0, stream>>>(Wq, wq, w4);
        cvt_f32_f16<<<(w4 + 255) / 256, 256, 0, stream>>>(Wk, wk, w4);
        cvt_f32_f16<<<(w4 + 255) / 256, 256, 0, stream>>>(Wv, wv, w4);
        cvt_f32_f16<<<(w4 + 255) / 256, 256, 0, stream>>>(Wp, wp, w4);
    }

    dim3 ggrid(CC / 128, MM / 128);           // (8, 32)
    gemm_wmma<1><<<ggrid, 256, 0, stream>>>(xb, wq, bq, nullptr, qr);  // Q + RoPE
    gemm_wmma<1><<<ggrid, 256, 0, stream>>>(xb, wk, bk, nullptr, kr);  // K + RoPE
    gemm_wmma<2><<<ggrid, 256, 0, stream>>>(xb, wv, bv, nullptr, vr);  // V

    attn_flash<<<dim3(TT / 64, BB * HH), 128, 0, stream>>>(qr, kr, vr, yb);

    gemm_wmma<0><<<ggrid, 256, 0, stream>>>(yb, wp, bp, out, nullptr); // projection
}
